// Seq2Seq_64742337019983
// MI455X (gfx1250) — matrix operand order repaired and checked by reading, not compiled, not run on hardware
//
#include <hip/hip_runtime.h>
#include <hip/hip_bf16.h>

typedef __attribute__((ext_vector_type(16))) _Float16 v16h;
typedef __attribute__((ext_vector_type(8)))  float    v8f;

namespace {
constexpr int kB = 1024;   // batch
constexpr int kS = 64;     // src len
constexpr int kH = 128;    // hidden
constexpr int kG = 512;    // 4*H (gates)
constexpr int kT = 24;     // output steps
}

// ---------- fast transcendentals: CDNA5 has hardware v_tanh_f32 ----------
__device__ __forceinline__ float fast_tanhf(float x) {
#if __has_builtin(__builtin_amdgcn_tanhf)
  return __builtin_amdgcn_tanhf(x);
#elif __has_builtin(__builtin_amdgcn_tanh_f32)
  return __builtin_amdgcn_tanh_f32(x);
#else
  return tanhf(x);
#endif
}
__device__ __forceinline__ float sigm(float x) {
#if __has_builtin(__builtin_amdgcn_tanhf) || __has_builtin(__builtin_amdgcn_tanh_f32)
  return 0.5f * fast_tanhf(0.5f * x) + 0.5f;   // 1 TRANS + 2 VALU
#else
  return 1.0f / (1.0f + __expf(-x));
#endif
}

// ---------- WMMA fragment loaders (ISA 7.12.2 layouts, wave32) ----------
// A: 16x32 f16. lanes 0-15: M=lane, halves 0..7 -> K=0..7, 8..15 -> K=16..23
//               lanes 16-31: M=lane-16, K=8..15 and 24..31
__device__ __forceinline__ v16h load_a_frag(const _Float16* __restrict__ A, int lda,
                                            int row0, int k0, int lane) {
  const int m  = lane & 15;
  const int hk = (lane & 16) ? 8 : 0;
  const _Float16* p = A + (size_t)(row0 + m) * lda + k0 + hk;
  union { uint4 u; _Float16 h[8]; } lo, hi;
  lo.u = *reinterpret_cast<const uint4*>(p);
  hi.u = *reinterpret_cast<const uint4*>(p + 16);
  v16h a;
#pragma unroll
  for (int j = 0; j < 8; ++j) { a[j] = lo.h[j]; a[8 + j] = hi.h[j]; }
  return a;
}

// B: 32x16 (K x N) from row-major W[N,K].  lanes 0-15: N=lane, K=0..7,16..23;
// lanes 16-31: N=lane-16, K=8..15,24..31
__device__ __forceinline__ v16h load_b_frag(const _Float16* __restrict__ W, int ldw,
                                            int n0, int k0, int lane) {
  const int n  = n0 + (lane & 15);
  const int kk = k0 + ((lane & 16) ? 8 : 0);
  const _Float16* p = W + (size_t)n * ldw + kk;
  union { uint4 u; _Float16 h[8]; } lo, hi;
  lo.u = *reinterpret_cast<const uint4*>(p);
  hi.u = *reinterpret_cast<const uint4*>(p + 16);
  v16h b;
#pragma unroll
  for (int j = 0; j < 8; ++j) { b[j] = lo.h[j]; b[8 + j] = hi.h[j]; }
  return b;
}

__device__ __forceinline__ v8f wmma_f16(v16h a, v16h b, v8f c) {
  return __builtin_amdgcn_wmma_f32_16x16x32_f16(false, a, false, b, (short)0, c,
                                                false, false);
}

// Gate GEMM for one 16-row tile: acc[g] += hrec @ Whh_g^T (+ xin @ Wih_g^T).
__device__ __forceinline__ void lstm_gates(const _Float16* hrec, const _Float16* xin,
                                           const _Float16* __restrict__ Whh,
                                           const _Float16* __restrict__ Wih,
                                           int w, int lane, v8f acc[4]) {
#pragma unroll
  for (int kb = 0; kb < kH; kb += 32) {
    const v16h ah = load_a_frag(hrec, kH, 0, kb, lane);
#pragma unroll
    for (int g = 0; g < 4; ++g)
      acc[g] = wmma_f16(ah, load_b_frag(Whh, kH, g * kH + w * 16, kb, lane), acc[g]);
    if (xin) {
      const v16h ax = load_a_frag(xin, kH, 0, kb, lane);
#pragma unroll
      for (int g = 0; g < 4; ++g)
        acc[g] = wmma_f16(ax, load_b_frag(Wih, kH, g * kH + w * 16, kb, lane), acc[g]);
    }
  }
}

// pointwise LSTM update; bias added here so WMMA chains start from inline-0 C.
__device__ __forceinline__ void lstm_pointwise(const v8f acc[4],
                                               const float* __restrict__ bias, int col,
                                               float c[8], float hn[8]) {
  const float bi = bias[0 * kH + col], bf = bias[1 * kH + col];
  const float bg = bias[2 * kH + col], bo = bias[3 * kH + col];
#pragma unroll
  for (int r = 0; r < 8; ++r) {
    const float ig = sigm(acc[0][r] + bi);
    const float fg = sigm(acc[1][r] + bf);
    const float gg = fast_tanhf(acc[2][r] + bg);
    const float og = sigm(acc[3][r] + bo);
    c[r]  = fg * c[r] + ig * gg;
    hn[r] = og * fast_tanhf(c[r]);
  }
}

// ================= fully fused encoder: 64 timesteps x 3 layers =================
__global__ __launch_bounds__(256) void encoder_kernel(
    const float* __restrict__ x,          // [B,S,2]
    const float* __restrict__ Wih0s,      // [512,2] f32
    const _Float16* __restrict__ Whh0,
    const _Float16* __restrict__ Wih1, const _Float16* __restrict__ Whh1,
    const _Float16* __restrict__ Wih2, const _Float16* __restrict__ Whh2,
    const float* __restrict__ b0, const float* __restrict__ b12,
    _Float16* __restrict__ enc_out,       // [B,S,128] f16
    _Float16* __restrict__ h0g, _Float16* __restrict__ h1g, _Float16* __restrict__ h2g,
    float* __restrict__ c0g, float* __restrict__ c1g, float* __restrict__ c2g)
{
  __shared__ __align__(16) _Float16 sh[3][16][kH];
  const int lane = threadIdx.x & 31;
  const int w    = threadIdx.x >> 5;
  const int b0r  = blockIdx.x * 16;
  const int n    = lane & 15;
  const int col  = w * 16 + n;
  const int rofs = (lane & 16) ? 8 : 0;

  float cst[3][8];
#pragma unroll
  for (int l = 0; l < 3; ++l)
#pragma unroll
    for (int r = 0; r < 8; ++r) cst[l][r] = 0.0f;
  for (int i = threadIdx.x; i < 3 * 16 * kH; i += 256)
    (&sh[0][0][0])[i] = (_Float16)0.0f;
  __syncthreads();

  for (int t = 0; t < kS; ++t) {
    // ---------- layer 0 (recurrent WMMA + tiny K=2 input) ----------
    v8f acc[4] = {};
    lstm_gates(&sh[0][0][0], nullptr, Whh0, nullptr, w, lane, acc);
    {
      float x0v[8], x1v[8];
#pragma unroll
      for (int r = 0; r < 8; ++r) {
        const size_t b = (size_t)(b0r + r + rofs);
        x0v[r] = x[b * (kS * 2) + t * 2 + 0];
        x1v[r] = x[b * (kS * 2) + t * 2 + 1];
      }
#pragma unroll
      for (int g = 0; g < 4; ++g) {
        const float w0 = Wih0s[(g * kH + col) * 2 + 0];
        const float w1 = Wih0s[(g * kH + col) * 2 + 1];
#pragma unroll
        for (int r = 0; r < 8; ++r) acc[g][r] += x0v[r] * w0 + x1v[r] * w1;
      }
    }
    float hn[8];
    lstm_pointwise(acc, b0, col, cst[0], hn);
    __syncthreads();
#pragma unroll
    for (int r = 0; r < 8; ++r) sh[0][r + rofs][col] = (_Float16)hn[r];
    __syncthreads();

    // ---------- layer 1 ----------
    v8f acc1[4] = {};
    lstm_gates(&sh[1][0][0], &sh[0][0][0], Whh1, Wih1, w, lane, acc1);
    lstm_pointwise(acc1, b12, col, cst[1], hn);
    __syncthreads();
#pragma unroll
    for (int r = 0; r < 8; ++r) sh[1][r + rofs][col] = (_Float16)hn[r];
    __syncthreads();

    // ---------- layer 2 (writes enc_out) ----------
    v8f acc2[4] = {};
    lstm_gates(&sh[2][0][0], &sh[1][0][0], Whh2, Wih2, w, lane, acc2);
    lstm_pointwise(acc2, b12 + kG, col, cst[2], hn);
    __syncthreads();
#pragma unroll
    for (int r = 0; r < 8; ++r) {
      const _Float16 hv = (_Float16)hn[r];
      sh[2][r + rofs][col] = hv;
      enc_out[((size_t)(b0r + r + rofs) * kS + t) * kH + col] = hv;
    }
    __syncthreads();
  }

  // ---- epilogue: persist final states for the decoder ----
#pragma unroll
  for (int r = 0; r < 8; ++r) {
    const size_t b = (size_t)(b0r + r + rofs);
    h0g[b * kH + col] = sh[0][r + rofs][col];
    h1g[b * kH + col] = sh[1][r + rofs][col];
    h2g[b * kH + col] = sh[2][r + rofs][col];
    c0g[b * kH + col] = cst[0][r];
    c1g[b * kH + col] = cst[1][r];
    c2g[b * kH + col] = cst[2][r];
  }
}

// ================= eproj = enc_out @ Wa_e^T + attn_b (wide one-shot GEMM) ========
__global__ __launch_bounds__(256) void gemm_eproj_kernel(
    const _Float16* __restrict__ A,        // [B*S,128]
    const _Float16* __restrict__ W,        // [128,128]
    const float* __restrict__ bias,        // [128]
    _Float16* __restrict__ D)              // [B*S,128]
{
  const int lane = threadIdx.x & 31;
  const int w    = threadIdx.x >> 5;
  const int m0   = blockIdx.x * 16;
  const int n0   = w * 16;
  const int n    = n0 + (lane & 15);
  const int rofs = (lane & 16) ? 8 : 0;

  v8f acc = {};
#pragma unroll
  for (int kb = 0; kb < kH; kb += 32)
    acc = wmma_f16(load_a_frag(A, kH, m0, kb, lane),
                   load_b_frag(W, kH, n0, kb, lane), acc);
  const float bv = bias[n];
#pragma unroll
  for (int r = 0; r < 8; ++r)
    D[(size_t)(m0 + r + rofs) * kH + n] = (_Float16)(acc[r] + bv);
}

// ================= fully fused decoder: 24 steps in one launch ===================
__global__ __launch_bounds__(256) void decoder_kernel(
    const float* __restrict__ prev_y,      // [B,2]
    const _Float16* __restrict__ enc_out,  // [B,S,128]
    const _Float16* __restrict__ eproj,    // [B,S,128]
    const _Float16* __restrict__ Wah,      // [128,128] (h_top half of attn_W)
    const float* __restrict__ vW,          // [128]
    const float* __restrict__ Wih0s,       // dec_Wih0 f32 [512,130] (cols 0..1)
    const _Float16* __restrict__ Wih0b,    // dec_Wih0 cols 2..129, f16 [512,128]
    const _Float16* __restrict__ Whh0,
    const _Float16* __restrict__ Wih1, const _Float16* __restrict__ Whh1,
    const _Float16* __restrict__ Wih2, const _Float16* __restrict__ Whh2,
    const float* __restrict__ b0v, const float* __restrict__ b12,
    const _Float16* __restrict__ h0g, const _Float16* __restrict__ h1g,
    const _Float16* __restrict__ h2g,
    const float* __restrict__ c0g, const float* __restrict__ c1g,
    const float* __restrict__ c2g,
    const float* __restrict__ outW,        // [2,256]
    const float* __restrict__ outb,        // [2]
    float* __restrict__ out)               // [T,B,2]
{
  __shared__ __align__(16) _Float16 sh[3][16][kH];   // decoder h state
  __shared__ __align__(16) _Float16 shw16[16][kH];   // weighted (f16, WMMA A input)
  __shared__ float sh_hproj[16][kH];
  __shared__ float sh_w32[16][kH];
  __shared__ float sc[16][kS];
  __shared__ float svw[kH];
  __shared__ float sinv[16];
  __shared__ float sprev[16][2];

  const int tid  = threadIdx.x;
  const int lane = tid & 31;
  const int w    = tid >> 5;
  const int b0r  = blockIdx.x * 16;
  const int n    = lane & 15;
  const int col  = w * 16 + n;
  const int rofs = (lane & 16) ? 8 : 0;

  float cst[3][8];
#pragma unroll
  for (int r = 0; r < 8; ++r) {
    const size_t b = (size_t)(b0r + r + rofs);
    sh[0][r + rofs][col] = h0g[b * kH + col];
    sh[1][r + rofs][col] = h1g[b * kH + col];
    sh[2][r + rofs][col] = h2g[b * kH + col];
    cst[0][r] = c0g[b * kH + col];
    cst[1][r] = c1g[b * kH + col];
    cst[2][r] = c2g[b * kH + col];
  }
  if (tid < 32) {
    const int row = tid >> 1, k = tid & 1;
    sprev[row][k] = prev_y[(size_t)(b0r + row) * 2 + k];
  }
  if (tid < kH) svw[tid] = vW[tid];
  __syncthreads();

  for (int t = 0; t < kT; ++t) {
    // ---- hproj = h_top @ Wa_h^T (WMMA, A from LDS) ----
    {
      v8f acc = {};
#pragma unroll
      for (int kb = 0; kb < kH; kb += 32)
        acc = wmma_f16(load_a_frag(&sh[2][0][0], kH, 0, kb, lane),
                       load_b_frag(Wah, kH, w * 16, kb, lane), acc);
#pragma unroll
      for (int r = 0; r < 8; ++r) sh_hproj[r + rofs][col] = acc[r];
    }
    __syncthreads();

    // ---- scores: thread does 4 of 16x64 scores; hw tanh + b128 eproj loads ----
    {
      const int row = tid >> 4;
      const int tt0 = tid & 15;
#pragma unroll
      for (int j = 0; j < 4; ++j) {
        const int tt = tt0 + 16 * j;
        const _Float16* ep = eproj + ((size_t)(b0r + row) * kS + tt) * kH;
        float s = 0.0f;
        for (int h8 = 0; h8 < kH; h8 += 8) {
          union { uint4 u; _Float16 h[8]; } e;
          e.u = *reinterpret_cast<const uint4*>(ep + h8);
#pragma unroll
          for (int q = 0; q < 8; ++q)
            s += fast_tanhf(sh_hproj[row][h8 + q] + (float)e.h[q]) * svw[h8 + q];
        }
        sc[row][tt] = s;
      }
    }
    __syncthreads();

    // ---- softmax over S=64, one thread per row ----
    if (tid < 16) {
      const int row = tid;
      float m = sc[row][0];
      for (int s = 1; s < kS; ++s) m = fmaxf(m, sc[row][s]);
      float ssum = 0.0f;
      for (int s = 0; s < kS; ++s) {
        const float e = __expf(sc[row][s] - m);
        sc[row][s] = e;
        ssum += e;
      }
      sinv[row] = 1.0f / ssum;
    }
    __syncthreads();

    // ---- weighted sum: 8 contiguous h per thread, b128 enc_out loads ----
    {
      const int row = tid >> 4;
      const int h0c = (tid & 15) * 8;
      float accw[8];
#pragma unroll
      for (int q = 0; q < 8; ++q) accw[q] = 0.0f;
      for (int ss = 0; ss < kS; ++ss) {
        union { uint4 u; _Float16 h[8]; } e;
        e.u = *reinterpret_cast<const uint4*>(
            enc_out + ((size_t)(b0r + row) * kS + ss) * kH + h0c);
        const float a = sc[row][ss];
#pragma unroll
        for (int q = 0; q < 8; ++q) accw[q] += a * (float)e.h[q];
      }
      const float inv = sinv[row];
#pragma unroll
      for (int q = 0; q < 8; ++q) {
        const float s = accw[q] * inv;
        sh_w32[row][h0c + q] = s;
        shw16[row][h0c + q]  = (_Float16)s;
      }
    }
    __syncthreads();

    // ---- decoder layer 0: rnn_in = [prev_out(2) | weighted(128)] ----
    {
      v8f acc[4] = {};
      lstm_gates(&sh[0][0][0], &shw16[0][0], Whh0, Wih0b, w, lane, acc);
      float x0v[8], x1v[8];
#pragma unroll
      for (int r = 0; r < 8; ++r) {
        x0v[r] = sprev[r + rofs][0];
        x1v[r] = sprev[r + rofs][1];
      }
#pragma unroll
      for (int g = 0; g < 4; ++g) {
        const float w0 = Wih0s[(size_t)(g * kH + col) * (kH + 2) + 0];
        const float w1 = Wih0s[(size_t)(g * kH + col) * (kH + 2) + 1];
#pragma unroll
        for (int r = 0; r < 8; ++r) acc[g][r] += x0v[r] * w0 + x1v[r] * w1;
      }
      float hn[8];
      lstm_pointwise(acc, b0v, col, cst[0], hn);
      __syncthreads();
#pragma unroll
      for (int r = 0; r < 8; ++r) sh[0][r + rofs][col] = (_Float16)hn[r];
      __syncthreads();
    }
    // ---- decoder layer 1 ----
    {
      v8f acc[4] = {};
      lstm_gates(&sh[1][0][0], &sh[0][0][0], Whh1, Wih1, w, lane, acc);
      float hn[8];
      lstm_pointwise(acc, b12, col, cst[1], hn);
      __syncthreads();
#pragma unroll
      for (int r = 0; r < 8; ++r) sh[1][r + rofs][col] = (_Float16)hn[r];
      __syncthreads();
    }
    // ---- decoder layer 2 ----
    {
      v8f acc[4] = {};
      lstm_gates(&sh[2][0][0], &sh[1][0][0], Whh2, Wih2, w, lane, acc);
      float hn[8];
      lstm_pointwise(acc, b12 + kG, col, cst[2], hn);
      __syncthreads();
#pragma unroll
      for (int r = 0; r < 8; ++r) sh[2][r + rofs][col] = (_Float16)hn[r];
      __syncthreads();
    }

    // ---- output head ----
    if (tid < 32) {
      const int row = tid >> 1, k = tid & 1;
      float a = outb[k];
      for (int j = 0; j < kH; ++j) a += (float)sh[2][row][j] * outW[k * 256 + j];
      for (int j = 0; j < kH; ++j) a += sh_w32[row][j] * outW[k * 256 + kH + j];
      out[(size_t)t * kB * 2 + (size_t)(b0r + row) * 2 + k] = a;
      sprev[row][k] = a;
    }
    __syncthreads();
  }
}

// ---------- weight conversion f32 -> f16 with stride/offset extraction ----------
__global__ void cvt_f32_f16_kernel(const float* __restrict__ src, _Float16* __restrict__ dst,
                                   int rows, int cols, int ld, int off) {
  const int i = blockIdx.x * 256 + threadIdx.x;
  if (i >= rows * cols) return;
  const int r = i / cols, c = i - r * cols;
  dst[i] = (_Float16)src[(size_t)r * ld + off + c];
}

extern "C" void kernel_launch(void* const* d_in, const int* in_sizes, int n_in,
                              void* d_out, int out_size, void* d_ws, size_t ws_size,
                              hipStream_t stream) {
  (void)in_sizes; (void)n_in; (void)out_size; (void)ws_size;
  const float* x         = (const float*)d_in[0];
  const float* prev_y    = (const float*)d_in[1];
  const float* enc_Wih0  = (const float*)d_in[2];
  const float* enc_Whh0  = (const float*)d_in[3];
  const float* enc_b0    = (const float*)d_in[4];
  const float* enc_Wih12 = (const float*)d_in[5];
  const float* enc_Whh12 = (const float*)d_in[6];
  const float* enc_b12   = (const float*)d_in[7];
  const float* attn_W    = (const float*)d_in[8];
  const float* attn_b    = (const float*)d_in[9];
  const float* v_W       = (const float*)d_in[10];
  const float* dec_Wih0  = (const float*)d_in[11];
  const float* dec_Whh0  = (const float*)d_in[12];
  const float* dec_b0    = (const float*)d_in[13];
  const float* dec_Wih12 = (const float*)d_in[14];
  const float* dec_Whh12 = (const float*)d_in[15];
  const float* dec_b12   = (const float*)d_in[16];
  const float* out_W     = (const float*)d_in[17];
  const float* out_b     = (const float*)d_in[18];
  float* out = (float*)d_out;

  char* wsb = (char*)d_ws;
  size_t off = 0;
  auto alloc = [&](size_t bytes) -> void* {
    off = (off + 255) & ~(size_t)255;
    void* p = wsb + off;
    off += bytes;
    return p;
  };

  const size_t wgt = (size_t)kG * kH * sizeof(_Float16);
  _Float16* w_enc_whh0  = (_Float16*)alloc(wgt);
  _Float16* w_enc_wih1  = (_Float16*)alloc(wgt);
  _Float16* w_enc_wih2  = (_Float16*)alloc(wgt);
  _Float16* w_enc_whh1  = (_Float16*)alloc(wgt);
  _Float16* w_enc_whh2  = (_Float16*)alloc(wgt);
  _Float16* w_dec_wih0b = (_Float16*)alloc(wgt);
  _Float16* w_dec_whh0  = (_Float16*)alloc(wgt);
  _Float16* w_dec_wih1  = (_Float16*)alloc(wgt);
  _Float16* w_dec_wih2  = (_Float16*)alloc(wgt);
  _Float16* w_dec_whh1  = (_Float16*)alloc(wgt);
  _Float16* w_dec_whh2  = (_Float16*)alloc(wgt);
  _Float16* w_attn_h    = (_Float16*)alloc((size_t)kH * kH * sizeof(_Float16));
  _Float16* w_attn_e    = (_Float16*)alloc((size_t)kH * kH * sizeof(_Float16));
  _Float16* enc_out = (_Float16*)alloc((size_t)kB * kS * kH * sizeof(_Float16));
  _Float16* eproj   = (_Float16*)alloc((size_t)kB * kS * kH * sizeof(_Float16));
  _Float16* h0g = (_Float16*)alloc((size_t)kB * kH * sizeof(_Float16));
  _Float16* h1g = (_Float16*)alloc((size_t)kB * kH * sizeof(_Float16));
  _Float16* h2g = (_Float16*)alloc((size_t)kB * kH * sizeof(_Float16));
  float* c0g = (float*)alloc((size_t)kB * kH * sizeof(float));
  float* c1g = (float*)alloc((size_t)kB * kH * sizeof(float));
  float* c2g = (float*)alloc((size_t)kB * kH * sizeof(float));

  // ---- weight prep: f32 -> f16 (with strided extraction) ----
  auto cvt = [&](const float* src, _Float16* dst, int rows, int cols, int ld, int o) {
    const int nelem = rows * cols;
    cvt_f32_f16_kernel<<<(nelem + 255) / 256, 256, 0, stream>>>(src, dst, rows, cols, ld, o);
  };
  cvt(enc_Whh0,                w_enc_whh0,  kG, kH, kH, 0);
  cvt(enc_Wih12 + 0 * kG * kH, w_enc_wih1,  kG, kH, kH, 0);
  cvt(enc_Wih12 + 1 * kG * kH, w_enc_wih2,  kG, kH, kH, 0);
  cvt(enc_Whh12 + 0 * kG * kH, w_enc_whh1,  kG, kH, kH, 0);
  cvt(enc_Whh12 + 1 * kG * kH, w_enc_whh2,  kG, kH, kH, 0);
  cvt(dec_Wih0,                w_dec_wih0b, kG, kH, kH + 2, 2);   // cols 2..129
  cvt(dec_Whh0,                w_dec_whh0,  kG, kH, kH, 0);
  cvt(dec_Wih12 + 0 * kG * kH, w_dec_wih1,  kG, kH, kH, 0);
  cvt(dec_Wih12 + 1 * kG * kH, w_dec_wih2,  kG, kH, kH, 0);
  cvt(dec_Whh12 + 0 * kG * kH, w_dec_whh1,  kG, kH, kH, 0);
  cvt(dec_Whh12 + 1 * kG * kH, w_dec_whh2,  kG, kH, kH, 0);
  cvt(attn_W,                  w_attn_h,    kH, kH, 2 * kH, 0);   // h_top half
  cvt(attn_W,                  w_attn_e,    kH, kH, 2 * kH, kH);  // enc_out half

  // ---- encoder: one launch, full 64-step recurrence, state in LDS/VGPRs ----
  encoder_kernel<<<kB / 16, 256, 0, stream>>>(
      x, enc_Wih0, w_enc_whh0, w_enc_wih1, w_enc_whh1, w_enc_wih2, w_enc_whh2,
      enc_b0, enc_b12, enc_out, h0g, h1g, h2g, c0g, c1g, c2g);

  // ---- eproj = enc_out @ Wa_e^T + attn_b (step-invariant attention half) ----
  gemm_eproj_kernel<<<(kB * kS) / 16, 256, 0, stream>>>(enc_out, w_attn_e, attn_b, eproj);

  // ---- decoder: one launch, full 24-step autoregression ----
  decoder_kernel<<<kB / 16, 256, 0, stream>>>(
      prev_y, enc_out, eproj, w_attn_h, v_W,
      dec_Wih0, w_dec_wih0b, w_dec_whh0, w_dec_wih1, w_dec_whh1, w_dec_wih2, w_dec_whh2,
      dec_b0, dec_b12, h0g, h1g, h2g, c0g, c1g, c2g, out_W, out_b, out);
}
